// SwinTransformerBlock_8985071583543
// MI455X (gfx1250) — matrix operand order repaired and checked by reading, not compiled, not run on hardware
//
#include <hip/hip_runtime.h>
#include <hip/hip_bf16.h>
#include <math.h>

// ---- Swin block constants --------------------------------------------------
#define C_DIM   128
#define IMG     56
#define NHEAD   4
#define HD      32
#define WSZ     7
#define SHFT    3
#define NTOK    49          // tokens per window
#define NWIN    64          // windows per image (8x8)
#define BATCH   64
#define HIDDEN  512

typedef __bf16 bf16x16 __attribute__((ext_vector_type(16)));
typedef __bf16 bf16x8  __attribute__((ext_vector_type(8)));
typedef float  f32x8   __attribute__((ext_vector_type(8)));

// ---- WMMA + fragment helpers (CDNA5 16x16x32 bf16 layouts, ISA 7.12.2) ----
__device__ __forceinline__ f32x8 wmma_bf16(bf16x16 a, bf16x16 b, f32x8 c) {
    return __builtin_amdgcn_wmma_f32_16x16x32_bf16(false, a, false, b,
                                                   (short)0, c, false, false);
}

__device__ __forceinline__ bf16x16 combine8(bf16x8 lo, bf16x8 hi) {
    bf16x16 r;
#pragma unroll
    for (int e = 0; e < 8; ++e) { r[e] = lo[e]; r[e + 8] = hi[e]; }
    return r;
}

// A fragment (16x32 MxK), row-major bf16 in LDS, base -> (m0,k0).
__device__ __forceinline__ bf16x16 load_a_lds(const __bf16* base, int ld, int lane) {
    const __bf16* p = base + (lane & 15) * ld + (lane >> 4) * 16;
    bf16x8 lo = *(const bf16x8*)p;
    bf16x8 hi = *(const bf16x8*)(p + 8);
    return combine8(lo, hi);
}

// B fragment (32x16 KxN) from an N-major source (B(k,n) = base[n*ld + k]).
__device__ __forceinline__ bf16x16 load_bT_lds(const __bf16* base, int ld, int lane) {
    const __bf16* p = base + (lane & 15) * ld + (lane >> 4) * 16;
    bf16x8 lo = *(const bf16x8*)p;
    bf16x8 hi = *(const bf16x8*)(p + 8);
    return combine8(lo, hi);
}

// B fragment from pre-packed global bf16 weights (one 32B load per lane).
__device__ __forceinline__ bf16x16 load_b_pk(const __bf16* __restrict__ Wp,
                                             int KT, int kt, int nt, int lane) {
    return reinterpret_cast<const bf16x16*>(Wp)[(nt * KT + kt) * 32 + lane];
}

// ---- Kernel 0: pack fp32 weight W[K][N] into B-fragment tile order --------
__global__ __launch_bounds__(256) void pack_w(const float* __restrict__ src,
                                              __bf16* __restrict__ dst,
                                              int K, int N) {
    const int tid = blockIdx.x * blockDim.x + threadIdx.x;
    if (tid >= K * N) return;
    const int KT   = K >> 5;
    const int tile = tid >> 9;
    const int r    = tid & 511;
    const int lane = r >> 4;
    const int q    = r & 15;
    const int nt   = tile / KT;
    const int kt   = tile - nt * KT;
    const int k = kt * 32 + (lane >> 4) * 16 + q;
    const int n = nt * 16 + (lane & 15);
    dst[tid] = (__bf16)src[k * N + n];
}

// ---- Kernel 1: LN1 + shifted-window attention + proj + residual -----------
// LDS ~122KB -> 2 workgroups per 320KB WGP (4 waves/SIMD).
__global__ __launch_bounds__(256) void swin_attn(
    const float* __restrict__ x,
    const float* __restrict__ g1, const float* __restrict__ b1,
    const __bf16* __restrict__ qkvp, const float* __restrict__ qkvb,
    const float* __restrict__ bias_tab,
    const __bf16* __restrict__ projp, const float* __restrict__ projb,
    float* __restrict__ y)
{
    __shared__ __align__(32) __bf16 sX[64 * 136];       // xnorm, reused for O
    __shared__ __align__(32) __bf16 sQK[64 * 264];      // q|k token-major
    __shared__ __align__(32) __bf16 sVt[C_DIM * 72];    // V channel-major
    __shared__ __align__(32) float  sS[2 * 64 * 66];    // logits (2 heads); f32 out
    __shared__ __align__(32) __bf16 sP[2 * 64 * 72];    // probabilities (2 heads)

    const int lane = threadIdx.x & 31;
    const int wid  = __builtin_amdgcn_readfirstlane(threadIdx.x >> 5);
    const int half = lane >> 4;
    const int b    = blockIdx.x >> 6;
    const int widx = blockIdx.x & 63;
    const int wy = widx >> 3, wx = widx & 7;
    const float* __restrict__ xb = x + (size_t)b * 3136 * 128;
    float* __restrict__ yb       = y + (size_t)b * 3136 * 128;

    // ---- Stage 1: LayerNorm + cyclic-shift window gather -> sX (bf16) ----
    for (int it = 0; it < 8; ++it) {
        const int t = wid + it * 8;            // 0..63 (wave-uniform)
        if (t < NTOK) {
            const int i = t / 7, j = t % 7;
            int h = wy * 7 + i + SHFT; if (h >= IMG) h -= IMG;
            int w = wx * 7 + j + SHFT; if (w >= IMG) w -= IMG;
            const float4 v = reinterpret_cast<const float4*>(xb + (h * 56 + w) * 128)[lane];
            float s  = v.x + v.y + v.z + v.w;
            float ss = v.x * v.x + v.y * v.y + v.z * v.z + v.w * v.w;
#pragma unroll
            for (int off = 16; off; off >>= 1) {
                s  += __shfl_xor(s, off, 32);
                ss += __shfl_xor(ss, off, 32);
            }
            const float mu   = s * (1.f / 128.f);
            const float rstd = rsqrtf(ss * (1.f / 128.f) - mu * mu + 1e-5f);
            const float vv[4] = {v.x, v.y, v.z, v.w};
#pragma unroll
            for (int c = 0; c < 4; ++c) {
                const int ch = lane * 4 + c;
                sX[t * 136 + ch] = (__bf16)((vv[c] - mu) * rstd * g1[ch] + b1[ch]);
            }
        } else {
#pragma unroll
            for (int c = 0; c < 4; ++c) sX[t * 136 + lane * 4 + c] = (__bf16)0.f;
        }
    }
    __syncthreads();

    // ---- Stage 2: QKV GEMM (64x128 @ 128x384) ----
    for (int it = 0; it < 12; ++it) {
        const int tid = __builtin_amdgcn_readfirstlane(wid + it * 8);  // 0..95
        const int mt = tid / 24, nt = tid % 24;
        f32x8 acc = {};
#pragma unroll
        for (int kk = 0; kk < 4; ++kk) {
            if (kk < 3)
                __builtin_prefetch((const void*)(reinterpret_cast<const bf16x16*>(qkvp)
                                   + (nt * 4 + kk + 1) * 32 + lane), 0, 3);
            bf16x16 a  = load_a_lds(sX + mt * 16 * 136 + kk * 32, 136, lane);
            bf16x16 bm = load_b_pk(qkvp, 4, kk, nt, lane);
            acc = wmma_bf16(a, bm, acc);
        }
        const int n      = nt * 16 + (lane & 15);
        const float bias = qkvb[n];
        if (nt < 16) {                          // scalar (SGPR) branch: Q|K
#pragma unroll
            for (int v = 0; v < 8; ++v) {
                const int m = mt * 16 + v + half * 8;
                sQK[m * 264 + n] = (__bf16)(acc[v] + bias);
            }
        } else {                                // V slice: m contiguous per half
            bf16x8 pk;
#pragma unroll
            for (int v = 0; v < 8; ++v) pk[v] = (__bf16)(acc[v] + bias);
            *(bf16x8*)(sVt + (n - 256) * 72 + mt * 16 + half * 8) = pk;
        }
    }
    __syncthreads();

    // ---- Stages 3-5: attention, two heads at a time (halves sS/sP LDS) ----
    for (int hp = 0; hp < 2; ++hp) {
        // S = Q @ K^T  (2 heads x 4x4 tiles = 32)
        for (int it = 0; it < 4; ++it) {
            const int tid = __builtin_amdgcn_readfirstlane(wid + it * 8); // 0..31
            const int hl = tid >> 4, mt = (tid >> 2) & 3, nt = tid & 3;
            const int hd = hp * 2 + hl;
            f32x8 acc = {};
            bf16x16 a  = load_a_lds(sQK + mt * 16 * 264 + hd * 32, 264, lane);
            bf16x16 bm = load_bT_lds(sQK + (nt * 16) * 264 + 128 + hd * 32, 264, lane);
            acc = wmma_bf16(a, bm, acc);
            const int n = nt * 16 + (lane & 15);
#pragma unroll
            for (int v = 0; v < 8; ++v) {
                const int m = mt * 16 + v + half * 8;
                sS[(hl * 64 + m) * 66 + n] = acc[v];
            }
        }
        __syncthreads();

        // softmax (+scale, rel-pos bias, shift mask) -> sP
        {
            const int r = threadIdx.x;
            if (r < 2 * NTOK) {
                const int hl = r / NTOK, n = r % NTOK;
                const int hd = hp * 2 + hl;
                float*  srow = sS + (hl * 64 + n) * 66;
                __bf16* prow = sP + (hl * 64 + n) * 72;
                const int i1 = n / 7, j1 = n % 7;
                const int h1 = wy * 7 + i1, w1 = wx * 7 + j1;
                const int reg1 = (h1 < 49 ? 0 : (h1 < 53 ? 1 : 2)) * 3 +
                                 (w1 < 49 ? 0 : (w1 < 53 ? 1 : 2));
                float mx = -1e30f;
                for (int m = 0; m < NTOK; ++m) {
                    const int i2 = m / 7, j2 = m % 7;
                    const int h2 = wy * 7 + i2, w2 = wx * 7 + j2;
                    const int reg2 = (h2 < 49 ? 0 : (h2 < 53 ? 1 : 2)) * 3 +
                                     (w2 < 49 ? 0 : (w2 < 53 ? 1 : 2));
                    const int rel = (i1 - i2 + 6) * 13 + (j1 - j2 + 6);
                    float s = srow[m] * 0.17677669529663687f
                            + bias_tab[rel * NHEAD + hd]
                            + ((reg1 != reg2) ? -100.f : 0.f);
                    srow[m] = s;
                    mx = fmaxf(mx, s);
                }
                float sum = 0.f;
                for (int m = 0; m < NTOK; ++m) {
                    const float e = __expf(srow[m] - mx);
                    srow[m] = e;
                    sum += e;
                }
                const float inv = 1.f / sum;
                for (int m = 0; m < NTOK; ++m) prow[m] = (__bf16)(srow[m] * inv);
                for (int m = NTOK; m < 64; ++m) prow[m] = (__bf16)0.f;
            }
        }
        __syncthreads();

        // O = P @ V  (2 heads x 4x2 tiles = 16), O -> sX (reuse)
        for (int it = 0; it < 2; ++it) {
            const int tid = __builtin_amdgcn_readfirstlane(wid + it * 8); // 0..15
            const int hl = tid >> 3, mt = (tid >> 1) & 3, nt = tid & 1;
            const int hd = hp * 2 + hl;
            f32x8 acc = {};
#pragma unroll
            for (int kk = 0; kk < 2; ++kk) {
                bf16x16 a  = load_a_lds(sP + (hl * 64 + mt * 16) * 72 + kk * 32, 72, lane);
                bf16x16 bm = load_bT_lds(sVt + (hd * 32 + nt * 16) * 72 + kk * 32, 72, lane);
                acc = wmma_bf16(a, bm, acc);
            }
            const int n = hd * 32 + nt * 16 + (lane & 15);
#pragma unroll
            for (int v = 0; v < 8; ++v) {
                const int m = mt * 16 + v + half * 8;
                sX[m * 136 + n] = (__bf16)acc[v];
            }
        }
        __syncthreads();
    }

    // ---- Stage 6: proj GEMM -> f32 tile in LDS (reuse sS: 64x132 fits) ----
    float* sOutF = sS;
    for (int it = 0; it < 4; ++it) {
        const int tid = __builtin_amdgcn_readfirstlane(wid + it * 8);    // 0..31
        const int mt = tid >> 3, nt = tid & 7;
        f32x8 acc = {};
#pragma unroll
        for (int kk = 0; kk < 4; ++kk) {
            bf16x16 a  = load_a_lds(sX + mt * 16 * 136 + kk * 32, 136, lane);
            bf16x16 bm = load_b_pk(projp, 4, kk, nt, lane);
            acc = wmma_bf16(a, bm, acc);
        }
        const int n    = nt * 16 + (lane & 15);
        const float pb = projb[n];
#pragma unroll
        for (int v = 0; v < 8; ++v) {
            const int m = mt * 16 + v + half * 8;
            sOutF[m * 132 + n] = acc[v] + pb;
        }
    }
    __syncthreads();

    // ---- Stage 7: coalesced residual add + scatter to un-shifted rows ----
    for (int it = 0; it < 8; ++it) {
        const int t = wid + it * 8;
        if (t < NTOK) {
            const int i = t / 7, j = t % 7;
            int h = wy * 7 + i + SHFT; if (h >= IMG) h -= IMG;
            int w = wx * 7 + j + SHFT; if (w >= IMG) w -= IMG;
            const size_t row = (size_t)(h * 56 + w) * 128;
            const float4 xv = reinterpret_cast<const float4*>(xb + row)[lane];
            const float4 ov = reinterpret_cast<const float4*>(sOutF + t * 132)[lane];
            float4 r;
            r.x = xv.x + ov.x; r.y = xv.y + ov.y;
            r.z = xv.z + ov.z; r.w = xv.w + ov.w;
            reinterpret_cast<float4*>(yb + row)[lane] = r;
        }
    }
}

// ---- Kernel 2: LN2 + MLP (128->512 GELU ->128) + residual, in-place ------
__global__ __launch_bounds__(256) void swin_mlp(
    float* __restrict__ y,
    const float* __restrict__ g2, const float* __restrict__ b2,
    const __bf16* __restrict__ fc1p, const float* __restrict__ fc1b,
    const __bf16* __restrict__ fc2p, const float* __restrict__ fc2b)
{
    __shared__ __align__(32) __bf16 sX[64 * 136];
    __shared__ __align__(32) __bf16 sH[64 * 520];
    __shared__ __align__(32) float  sY[64 * 132];
    __shared__ __align__(32) float  sOut[64 * 132];

    const int lane = threadIdx.x & 31;
    const int wid  = __builtin_amdgcn_readfirstlane(threadIdx.x >> 5);
    const int half = lane >> 4;
    float* __restrict__ yb = y + (size_t)blockIdx.x * 64 * 128;

    // LN2 + stash residual
    for (int it = 0; it < 8; ++it) {
        const int t = wid + it * 8;
        const float4 v = reinterpret_cast<const float4*>(yb + t * 128)[lane];
        float s  = v.x + v.y + v.z + v.w;
        float ss = v.x * v.x + v.y * v.y + v.z * v.z + v.w * v.w;
#pragma unroll
        for (int off = 16; off; off >>= 1) {
            s  += __shfl_xor(s, off, 32);
            ss += __shfl_xor(ss, off, 32);
        }
        const float mu   = s * (1.f / 128.f);
        const float rstd = rsqrtf(ss * (1.f / 128.f) - mu * mu + 1e-5f);
        const float vv[4] = {v.x, v.y, v.z, v.w};
#pragma unroll
        for (int c = 0; c < 4; ++c) {
            const int ch = lane * 4 + c;
            sY[t * 132 + ch] = vv[c];
            sX[t * 136 + ch] = (__bf16)((vv[c] - mu) * rstd * g2[ch] + b2[ch]);
        }
    }
    __syncthreads();

    // fc1 + exact GELU (64x128 @ 128x512)
    for (int it = 0; it < 16; ++it) {
        const int tid = __builtin_amdgcn_readfirstlane(wid + it * 8);    // 0..127
        const int mt = tid >> 5, nt = tid & 31;
        f32x8 acc = {};
#pragma unroll
        for (int kk = 0; kk < 4; ++kk) {
            if (kk < 3)
                __builtin_prefetch((const void*)(reinterpret_cast<const bf16x16*>(fc1p)
                                   + (nt * 4 + kk + 1) * 32 + lane), 0, 3);
            bf16x16 a  = load_a_lds(sX + mt * 16 * 136 + kk * 32, 136, lane);
            bf16x16 bm = load_b_pk(fc1p, 4, kk, nt, lane);
            acc = wmma_bf16(a, bm, acc);
        }
        const int n    = nt * 16 + (lane & 15);
        const float fb = fc1b[n];
#pragma unroll
        for (int v = 0; v < 8; ++v) {
            const int m = mt * 16 + v + half * 8;
            float h = acc[v] + fb;
            h = 0.5f * h * (1.f + erff(h * 0.70710678118654752f));
            sH[m * 520 + n] = (__bf16)h;
        }
    }
    __syncthreads();

    // fc2 -> f32 tile in LDS (64x512 @ 512x128)
    for (int it = 0; it < 4; ++it) {
        const int tid = __builtin_amdgcn_readfirstlane(wid + it * 8);    // 0..31
        const int mt = tid >> 3, nt = tid & 7;
        f32x8 acc = {};
#pragma unroll
        for (int kk = 0; kk < 16; ++kk) {
            bf16x16 a  = load_a_lds(sH + mt * 16 * 520 + kk * 32, 520, lane);
            bf16x16 bm = load_b_pk(fc2p, 16, kk, nt, lane);
            acc = wmma_bf16(a, bm, acc);
        }
        const int n    = nt * 16 + (lane & 15);
        const float fb = fc2b[n];
#pragma unroll
        for (int v = 0; v < 8; ++v) {
            const int m = mt * 16 + v + half * 8;
            sOut[m * 132 + n] = acc[v] + fb;
        }
    }
    __syncthreads();

    // coalesced residual add + store
    for (int it = 0; it < 8; ++it) {
        const int t = wid + it * 8;
        const float4 rv = reinterpret_cast<const float4*>(sY + t * 132)[lane];
        const float4 ov = reinterpret_cast<const float4*>(sOut + t * 132)[lane];
        float4 r;
        r.x = rv.x + ov.x; r.y = rv.y + ov.y;
        r.z = rv.z + ov.z; r.w = rv.w + ov.w;
        reinterpret_cast<float4*>(yb + t * 128)[lane] = r;
    }
}

// ---- Host launch -----------------------------------------------------------
extern "C" void kernel_launch(void* const* d_in, const int* in_sizes, int n_in,
                              void* d_out, int out_size, void* d_ws, size_t ws_size,
                              hipStream_t stream) {
    (void)in_sizes; (void)n_in; (void)out_size; (void)ws_size;
    const float* x     = (const float*)d_in[0];
    const float* n1g   = (const float*)d_in[1];
    const float* n1b   = (const float*)d_in[2];
    const float* qkvw  = (const float*)d_in[3];
    const float* qkvb  = (const float*)d_in[4];
    const float* btab  = (const float*)d_in[5];
    const float* projw = (const float*)d_in[6];
    const float* projb = (const float*)d_in[7];
    const float* n2g   = (const float*)d_in[8];
    const float* n2b   = (const float*)d_in[9];
    const float* fc1w  = (const float*)d_in[10];
    const float* fc1b  = (const float*)d_in[11];
    const float* fc2w  = (const float*)d_in[12];
    const float* fc2b  = (const float*)d_in[13];
    float* out = (float*)d_out;

    // Packed bf16 weights in workspace (B-fragment tile order), 384 KB total.
    char* ws = (char*)d_ws;
    __bf16* qkvp  = (__bf16*)(ws + 0);        // 128x384 -> 98304 B
    __bf16* projp = (__bf16*)(ws + 98304);    // 128x128 -> 32768 B
    __bf16* fc1p  = (__bf16*)(ws + 131072);   // 128x512 -> 131072 B
    __bf16* fc2p  = (__bf16*)(ws + 262144);   // 512x128 -> 131072 B

    pack_w<<<(128 * 384 + 255) / 256, 256, 0, stream>>>(qkvw, qkvp, 128, 384);
    pack_w<<<(128 * 128 + 255) / 256, 256, 0, stream>>>(projw, projp, 128, 128);
    pack_w<<<(128 * 512 + 255) / 256, 256, 0, stream>>>(fc1w, fc1p, 128, 512);
    pack_w<<<(512 * 128 + 255) / 256, 256, 0, stream>>>(fc2w, fc2p, 512, 128);

    swin_attn<<<BATCH * NWIN, 256, 0, stream>>>(x, n1g, n1b, qkvp, qkvb, btab,
                                                projp, projb, out);
    swin_mlp<<<(BATCH * 3136) / 64, 256, 0, stream>>>(out, n2g, n2b,
                                                      fc1p, fc1b, fc2p, fc2b);
}
